// DSAutoCorrelation_11218454577567
// MI455X (gfx1250) — matrix operand order repaired and checked by reading, not compiled, not run on hardware
//
#include <hip/hip_runtime.h>
#include <hip/hip_bf16.h>
#include <math.h>

// ---------------------------------------------------------------------------
// AutoCorrelation attention (Autoformer) for MI455X / gfx1250.
// B=16, L=2048, H=8, E=64  (feature dim D = H*E = 512), TOPK = int(log L) = 7.
//
// Stage 1: mean_value[b,tau] = (1/512) * sum_t <q[b,(t+tau)%L], k[b,t]>
//          == circular-diagonal sums of G_b = Q_b * K_b^T  -> bf16 WMMA GEMM
// Stage 2: batch-summed top-7 + per-batch softmax over raw scores
// Stage 3: out[b,l,:] = sum_k w[b,k] * V[b,(l+idx_k)%L,:]
// ---------------------------------------------------------------------------

#define LSEQ 2048
#define DDIM 512
#define TOPK 7

typedef __attribute__((ext_vector_type(16))) __bf16 v16bf;
typedef __attribute__((ext_vector_type(8)))  __bf16 v8bf;
typedef __attribute__((ext_vector_type(8)))  float  v8f;

// ---------------------------------------------------------------------------
// Stage 0: zero the 16 x 2048 score accumulator in workspace
// ---------------------------------------------------------------------------
__global__ void zero_scores_kernel(float* mv) {
    mv[blockIdx.x * 256 + threadIdx.x] = 0.0f;
}

// ---------------------------------------------------------------------------
// Stage 1: G = Q K^T per batch, binned into circular diagonals.
// Grid: (L/64, L/128, B); block 256 threads = 8 waves.
// Macro-tile 128(M) x 64(N), K-depth 512 in chunks of 32 (bf16 WMMA).
// Wave w owns M-strip [16w, 16w+16) and all four 16-col tiles:
//   one A-fragment load feeds 4 WMMAs per K-chunk (4 accumulators).
// ---------------------------------------------------------------------------
__global__ __launch_bounds__(256) void corr_gemm_kernel(
        const float* __restrict__ Q, const float* __restrict__ K,
        float* __restrict__ mv) {
    const int b  = blockIdx.z;
    const int i0 = blockIdx.y * 128;    // G row block  (q index t+tau)
    const int j0 = blockIdx.x * 64;     // G col block  (k index t)

    // 40-element row stride keeps every 8-element chunk 16B aligned.
    __shared__ __align__(16) __bf16 At[128 * 40];
    __shared__ __align__(16) __bf16 Bt[64 * 40];
    __shared__ float diag[192];

    const int t    = threadIdx.x;
    const int lane = t & 31;
    const int wave = t >> 5;            // 0..7 : M-strip index

    v8f acc[4] = {v8f{}, v8f{}, v8f{}, v8f{}};

    const float* Qb = Q + (size_t)b * LSEQ * DDIM;
    const float* Kb = K + (size_t)b * LSEQ * DDIM;

    // Q tile fill: 128 rows x 32 cols, 16 f32 per thread (half a row)
    const int qrow = t >> 1;
    const int qcol = (t & 1) * 16;
    // K tile fill: 64 rows x 32 cols, 8 f32 per thread
    const int krow = t >> 2;
    const int kcol = (t & 3) * 8;

    // Fragment layout bases (ISA 7.12.2, wave32):
    //   A 16x32 bf16: lane m(=l%16) holds K = 8*(l>=16)+[0..7] and 16+8*(l>=16)+[0..7]
    //   B 32x16 bf16: lane n(=l%16) holds K = 8*(l>=16)+[0..7] and 16+8*(l>=16)+[0..7]
    const int m   = lane & 15;
    const int kbA = (lane >> 4) * 8;
    const int kbB = (lane >> 4) * 16;

    for (int k0 = 0; k0 < DDIM; k0 += 32) {
        // ---- global f32 -> LDS bf16 (native packed converts) ----
        {
            const float* qs = Qb + (size_t)(i0 + qrow) * DDIM + k0 + qcol;
            v8bf q0, q1;
            #pragma unroll
            for (int i = 0; i < 8; ++i) { q0[i] = (__bf16)qs[i]; q1[i] = (__bf16)qs[8 + i]; }
            *(v8bf*)&At[qrow * 40 + qcol]     = q0;
            *(v8bf*)&At[qrow * 40 + qcol + 8] = q1;

            const float* ks = Kb + (size_t)(j0 + krow) * DDIM + k0 + kcol;
            v8bf kk;
            #pragma unroll
            for (int i = 0; i < 8; ++i) kk[i] = (__bf16)ks[i];
            *(v8bf*)&Bt[krow * 40 + kcol] = kk;
        }
        __syncthreads();

        // ---- one A fragment per wave, reused across 4 B fragments ----
        v8bf alo = *(const v8bf*)&At[(wave * 16 + m) * 40 + kbA];
        v8bf ahi = *(const v8bf*)&At[(wave * 16 + m) * 40 + 16 + kbA];
        v16bf a = __builtin_shufflevector(alo, ahi,
                    0,1,2,3,4,5,6,7,8,9,10,11,12,13,14,15);

        #pragma unroll
        for (int c = 0; c < 4; ++c) {
            v8bf blo = *(const v8bf*)&Bt[(c * 16 + m) * 40 + kbA];
            v8bf bhi = *(const v8bf*)&Bt[(c * 16 + m) * 40 + 16 + kbA];
            v16bf bb = __builtin_shufflevector(blo, bhi,
                        0,1,2,3,4,5,6,7,8,9,10,11,12,13,14,15);
            acc[c] = __builtin_amdgcn_wmma_f32_16x16x32_bf16(
                         false, a, false, bb, (short)0, acc[c], false, false);
        }
        __syncthreads();
    }

    // ---- bin the 128x64 macro-tile into its 191 local diagonals ----
    if (t < 192) diag[t] = 0.0f;
    __syncthreads();

    // C/D layout: VGPR v, lanes 0-15: (M=v, N=lane); lanes 16-31: (M=v+8, N=lane-16)
    const int  Mbase = wave * 16 + ((lane >> 4) ? 8 : 0);
    const float scale = 1.0f / (float)DDIM;
    #pragma unroll
    for (int c = 0; c < 4; ++c) {
        const int N = c * 16 + m;
        #pragma unroll
        for (int v = 0; v < 8; ++v) {
            atomicAdd(&diag[Mbase + v - N + 63], acc[c][v] * scale);
        }
    }
    __syncthreads();

    if (t < 191) {
        const float val = diag[t];
        if (val != 0.0f) {
            const int tau = (i0 - j0 + (t - 63)) & (LSEQ - 1);
            atomicAdd(&mv[b * LSEQ + tau], val);
        }
    }
}

// ---------------------------------------------------------------------------
// Stage 2: single workgroup. Batch-sum scores, top-7 indices,
// per-batch softmax over the raw per-batch scores at those indices.
// ---------------------------------------------------------------------------
__global__ __launch_bounds__(256) void topk_softmax_kernel(
        const float* __restrict__ mv, int* __restrict__ idx_out,
        float* __restrict__ w_out) {
    __shared__ float s[LSEQ];
    __shared__ float rv[256];
    __shared__ int   ri[256];
    __shared__ int   idxs[TOPK + 1];

    const int t = threadIdx.x;

    for (int tau = t; tau < LSEQ; tau += 256) {
        float acc = 0.0f;
        #pragma unroll
        for (int b = 0; b < 16; ++b) acc += mv[b * LSEQ + tau];
        s[tau] = acc;
    }
    __syncthreads();

    for (int k = 0; k < TOPK; ++k) {
        float bv = -INFINITY; int bi = 0;
        for (int tau = t; tau < LSEQ; tau += 256) {
            if (s[tau] > bv) { bv = s[tau]; bi = tau; }
        }
        rv[t] = bv; ri[t] = bi;
        __syncthreads();
        for (int st = 128; st > 0; st >>= 1) {
            if (t < st && rv[t + st] > rv[t]) { rv[t] = rv[t + st]; ri[t] = ri[t + st]; }
            __syncthreads();
        }
        if (t == 0) { idxs[k] = ri[0]; s[ri[0]] = -INFINITY; }
        __syncthreads();
    }

    if (t < TOPK) idx_out[t] = idxs[t];

    if (t < 16) {   // one thread per batch: softmax over the 7 raw scores
        float w[TOPK];
        float mx = -INFINITY;
        #pragma unroll
        for (int k = 0; k < TOPK; ++k) {
            w[k] = mv[t * LSEQ + idxs[k]];
            mx = fmaxf(mx, w[k]);
        }
        float sum = 0.0f;
        #pragma unroll
        for (int k = 0; k < TOPK; ++k) { w[k] = __expf(w[k] - mx); sum += w[k]; }
        const float inv = 1.0f / sum;
        #pragma unroll
        for (int k = 0; k < TOPK; ++k) w_out[t * TOPK + k] = w[k] * inv;
    }
}

// ---------------------------------------------------------------------------
// Stage 3: out[b,l,:] = sum_k w[b,k] * V[b,(l+idx_k)%L,:]
// Grid (L, B); 128 threads, float4 per thread over the 512-float row.
// ---------------------------------------------------------------------------
__global__ __launch_bounds__(128) void aggregate_kernel(
        const float* __restrict__ V, const int* __restrict__ idxs,
        const float* __restrict__ w, float* __restrict__ out) {
    const int l = blockIdx.x;
    const int b = blockIdx.y;
    const int t = threadIdx.x;

    __shared__ int   sidx[TOPK];
    __shared__ float sw[TOPK];
    if (t < TOPK) { sidx[t] = idxs[t]; sw[t] = w[b * TOPK + t]; }
    __syncthreads();

    float4 acc = make_float4(0.f, 0.f, 0.f, 0.f);
    #pragma unroll
    for (int k = 0; k < TOPK; ++k) {
        const int row = (l + sidx[k]) & (LSEQ - 1);
        const float4 x = ((const float4*)(V + ((size_t)b * LSEQ + row) * DDIM))[t];
        const float wk = sw[k];
        acc.x += wk * x.x; acc.y += wk * x.y; acc.z += wk * x.z; acc.w += wk * x.w;
    }
    ((float4*)(out + ((size_t)b * LSEQ + l) * DDIM))[t] = acc;
}

// ---------------------------------------------------------------------------
extern "C" void kernel_launch(void* const* d_in, const int* in_sizes, int n_in,
                              void* d_out, int out_size, void* d_ws, size_t ws_size,
                              hipStream_t stream) {
    const float* Q = (const float*)d_in[0];
    const float* K = (const float*)d_in[1];
    const float* V = (const float*)d_in[2];
    // d_in[3] = attn_mask (unused by the reference math)

    float* mv   = (float*)d_ws;                                  // 16*2048 f32
    int*   idxs = (int*)((char*)d_ws + 16 * LSEQ * sizeof(float));
    float* w    = (float*)((char*)d_ws + 16 * LSEQ * sizeof(float) + 64);
    float* out  = (float*)d_out;

    zero_scores_kernel<<<(16 * LSEQ) / 256, 256, 0, stream>>>(mv);

    dim3 g1(LSEQ / 64, LSEQ / 128, 16);
    corr_gemm_kernel<<<g1, 256, 0, stream>>>(Q, K, mv);

    topk_softmax_kernel<<<1, 256, 0, stream>>>(mv, idxs, w);

    dim3 g3(LSEQ, 16);
    aggregate_kernel<<<g3, 128, 0, stream>>>(V, idxs, w, out);
}
